// MySHAN_79267916415237
// MI455X (gfx1250) — compile-verified, matrix operand order checked by reading, not run on hardware
//
#include <hip/hip_runtime.h>
#include <hip/hip_bf16.h>

// SHAN two-level attention for MI455X (gfx1250, wave32, WMMA + async-LDS).
// B=2048, N=9, L=50, D=64, H=16.
//
//  - pack kernel: W1/Wf1 (f32 [K=128][H=16]) -> f16 CDNA5 WMMA B-matrix
//    lane layout in d_ws.
//  - stage1: one 128-thread block per (b,n). Item embeddings gathered
//    directly into LDS with GLOBAL_LOAD_ASYNC_TO_LDS_B128 (ASYNCcnt),
//    rows 50..63 zero-filled so the A-pack is branch-free. 4 waves each run
//    a 16-row tile of relu(feat @ W1[n] + b1) via v_wmma_f32_16x16x32_f16
//    (K=128 in 4 chunks), logits via 16-lane shuffle butterfly, masked
//    softmax over L, weighted pooling -> u_area in d_ws.
//  - stage2: one wave per b; ue + 9 pooled rows staged in LDS (zero-padded
//    to 16 rows), same WMMA pattern, softmax over N, weighted sum -> d_out.

typedef __attribute__((ext_vector_type(16))) _Float16 v16h;
typedef __attribute__((ext_vector_type(8)))  float    v8f;

#define BN  9
#define LL  50
#define DD  64
#define HH  16

// d_ws layout (bytes):
//   [0, 36864)        : packed W1   (9 n * 4 chunks * 32 lanes * 16 halves)
//   [36864, 40960)    : packed Wf1  (4 chunks * 32 lanes * 16 halves)
//   [40960, +B*9*64*4): u_area f32
#define WS_UAREA_OFF 40960

// Branch-free pack of 16 f32 (two contiguous 8-float segments) -> v16h.
__device__ __forceinline__ v16h pack16(const float* s0, const float* s1) {
  float4 q0 = *(const float4*)(s0);
  float4 q1 = *(const float4*)(s0 + 4);
  float4 q2 = *(const float4*)(s1);
  float4 q3 = *(const float4*)(s1 + 4);
  v16h a;
  a[0]  = (_Float16)q0.x; a[1]  = (_Float16)q0.y;
  a[2]  = (_Float16)q0.z; a[3]  = (_Float16)q0.w;
  a[4]  = (_Float16)q1.x; a[5]  = (_Float16)q1.y;
  a[6]  = (_Float16)q1.z; a[7]  = (_Float16)q1.w;
  a[8]  = (_Float16)q2.x; a[9]  = (_Float16)q2.y;
  a[10] = (_Float16)q2.z; a[11] = (_Float16)q2.w;
  a[12] = (_Float16)q3.x; a[13] = (_Float16)q3.y;
  a[14] = (_Float16)q3.z; a[15] = (_Float16)q3.w;
  return a;
}

// Async 16-byte copy global -> LDS (no VGPR round-trip, tracked by ASYNCcnt).
// lds_off = wave-relative LDS byte offset (low 32 bits of a generic shared
// pointer are the LDS offset on gfx1250's shared aperture).
__device__ __forceinline__ void async_g2l_b128(unsigned lds_off,
                                               const void* gaddr) {
  asm volatile("global_load_async_to_lds_b128 %0, %1, off"
               :
               : "v"(lds_off), "v"(gaddr)
               : "memory");
}
__device__ __forceinline__ void wait_asynccnt0() {
  asm volatile("s_wait_asynccnt 0" ::: "memory");
}

// ---------------------------------------------------------------------------
// Pack W1[n]/Wf1 into CDNA5 16-bit B-matrix layout: lane holds column
// col=lane&15; element e holds K = 32*c + 8*g + e + (e>=8 ? 8 : 0), g=lane>>4.
// ---------------------------------------------------------------------------
__global__ void shan_pack(const float* __restrict__ W1,
                          const float* __restrict__ Wf1,
                          _Float16* __restrict__ dst) {
  int idx = blockIdx.x * blockDim.x + threadIdx.x;
  if (idx >= 40 * 32) return;               // 36 (n,c) pairs + 4 Wf1 chunks
  int pair = idx >> 5;
  int lane = idx & 31;
  int g = lane >> 4, col = lane & 15;
  const float* src;
  int c;
  if (pair < 36) { int n = pair >> 2; c = pair & 3; src = W1 + n * 128 * HH; }
  else           { c = pair - 36;               src = Wf1; }
  _Float16* o = dst + (long)idx * 16;
  #pragma unroll
  for (int e = 0; e < 16; ++e) {
    int K = 32 * c + 8 * g + e + (e >= 8 ? 8 : 0);
    o[e] = (_Float16)src[K * HH + col];
  }
}

// ---------------------------------------------------------------------------
// Stage 1: per-(b,n) list attention + pooling.
// ---------------------------------------------------------------------------
__global__ __launch_bounds__(128) void shan_stage1(
    const int*   __restrict__ user,
    const int*   __restrict__ item_ids,
    const int*   __restrict__ lengths,
    const float* __restrict__ user_table,
    const float* __restrict__ item_table,
    const float* __restrict__ b1,
    const float* __restrict__ W2,
    const float* __restrict__ b2,
    const v16h*  __restrict__ Bpack,
    float*       __restrict__ u_area) {
  const int blk = blockIdx.x;
  const int b = blk / BN, n = blk % BN;
  const int tid  = threadIdx.x;
  const int wave = tid >> 5, lane = tid & 31;
  const int g = lane >> 4, col = lane & 15;

  __shared__ float s_ue[DD];
  __shared__ float s_ae[64][DD + 4];        // 64 rows: 50 real + 14 zero pad
  __shared__ float s_logits[64];
  __shared__ float s_w[64];

  if (tid < DD) s_ue[tid] = user_table[(long)user[b] * DD + tid];

  // ---- gather item embeddings straight into LDS (async, B128 chunks) ----
  const int* ids = item_ids + ((long)b * BN + n) * LL;
  const unsigned ae_base = (unsigned)(unsigned long long)(void*)&s_ae[0][0];
  for (int idx = tid; idx < LL * (DD / 4); idx += 128) {
    int row = idx >> 4, q = idx & 15;
    const float* src = item_table + (long)ids[row] * DD + q * 4;
    async_g2l_b128(ae_base + (unsigned)(row * (DD + 4) + q * 4) * 4u, src);
  }
  // zero-fill padded rows 50..63 so the WMMA A-pack needs no row guard
  for (int idx = tid; idx < (64 - LL) * (DD / 4); idx += 128) {
    int row = LL + (idx >> 4), q = idx & 15;
    *(float4*)&s_ae[row][q * 4] = make_float4(0.f, 0.f, 0.f, 0.f);
  }
  const int len = lengths[b * BN + n];
  wait_asynccnt0();                          // async transfers visible in LDS
  __syncthreads();

  // --- WMMA: h = feat[16 rows x 128] @ W1[n] (accum f32) ------------------
  const int m0  = wave * 16;
  const int row = m0 + col;                 // A-matrix row (0..63, padded)

  v8f acc = {};
  // chunks 0..1: K in [0,64) -> user embedding (row-independent)
  #pragma unroll
  for (int c = 0; c < 2; ++c) {
    const int kb = 32 * c + 8 * g;
    v16h a  = pack16(&s_ue[kb], &s_ue[kb + 16]);
    v16h bm = Bpack[(n * 4 + c) * 32 + lane];
    acc = __builtin_amdgcn_wmma_f32_16x16x32_f16(
        false, a, false, bm, (short)0, acc, false, false);
  }
  // chunks 2..3: K in [64,128) -> item row (zero-padded rows are zeros)
  const float* rowp = &s_ae[row][0];
  #pragma unroll
  for (int c = 2; c < 4; ++c) {
    const int kb = 32 * (c - 2) + 8 * g;
    v16h a  = pack16(rowp + kb, rowp + kb + 16);
    v16h bm = Bpack[(n * 4 + c) * 32 + lane];
    acc = __builtin_amdgcn_wmma_f32_16x16x32_f16(
        false, a, false, bm, (short)0, acc, false, false);
  }

  // --- bias + ReLU, then logits = h . W2[n] via 16-lane butterfly ---------
  const float bias = b1[n * HH + col];
  const float w2c  = W2[n * HH + col];
  float p[8];
  #pragma unroll
  for (int r = 0; r < 8; ++r) {
    float v = acc[r] + bias;
    v = v > 0.0f ? v : 0.0f;
    p[r] = v * w2c;
  }
  #pragma unroll
  for (int mask = 1; mask <= 8; mask <<= 1) {
    #pragma unroll
    for (int r = 0; r < 8; ++r) p[r] += __shfl_xor(p[r], mask, 32);
  }
  if (col == 0) {
    const float b2v = b2[n];
    #pragma unroll
    for (int r = 0; r < 8; ++r) {
      int m = m0 + 8 * g + r;               // C/D layout: m = r + 8*g
      float lg = p[r] + b2v;
      if (m >= len) lg = -1e30f;            // mask invalid / padded rows
      s_logits[m] = lg;
    }
  }
  __syncthreads();

  // --- masked softmax over the 64 (padded) list slots ---------------------
  if (tid < 64) {
    float mx = -1e30f;
    for (int j = 0; j < 64; ++j) mx = fmaxf(mx, s_logits[j]);
    float sum = 0.0f;
    for (int j = 0; j < 64; ++j) sum += __expf(s_logits[j] - mx);
    s_w[tid] = __expf(s_logits[tid] - mx) / sum;
  }
  __syncthreads();

  // --- u_area[b,n,:] = sum_l w[l] * ae[l,:] -------------------------------
  if (tid < DD) {
    float s = 0.0f;
    for (int l = 0; l < LL; ++l) s += s_w[l] * s_ae[l][tid];
    u_area[((long)b * BN + n) * DD + tid] = s;
  }
}

// ---------------------------------------------------------------------------
// Stage 2: final attention over the N=9 pooled embeddings. One wave per b.
// ---------------------------------------------------------------------------
__global__ __launch_bounds__(32) void shan_stage2(
    const int*   __restrict__ user,
    const float* __restrict__ user_table,
    const float* __restrict__ bf1,
    const float* __restrict__ Wf2,
    const float* __restrict__ bf2,
    const v16h*  __restrict__ Bpack,
    const float* __restrict__ u_area,
    float*       __restrict__ out) {
  const int b = blockIdx.x;
  const int lane = threadIdx.x;
  const int g = lane >> 4, col = lane & 15;

  const float* ue = user_table + (long)user[b] * DD;
  const float* ua = u_area + (long)b * BN * DD;

  __shared__ float s_ue[DD];
  __shared__ float s_f[16][DD + 4];         // rows 0..8 = u_area, 9..15 zero
  __shared__ float s_lg[16];

  for (int d = lane; d < DD; d += 32) s_ue[d] = ue[d];
  for (int idx = lane; idx < BN * (DD / 4); idx += 32) {
    int r = idx >> 4, q = idx & 15;
    const float4 v = *(const float4*)(ua + r * DD + q * 4);
    *(float4*)&s_f[r][q * 4] = v;
  }
  for (int idx = lane; idx < (16 - BN) * (DD / 4); idx += 32) {
    int r = BN + (idx >> 4), q = idx & 15;
    *(float4*)&s_f[r][q * 4] = make_float4(0.f, 0.f, 0.f, 0.f);
  }
  __syncthreads();

  // --- WMMA: h2 = feat2[16(9) x 128] @ Wf1 --------------------------------
  v8f acc = {};
  #pragma unroll
  for (int c = 0; c < 2; ++c) {             // K in [0,64): user embedding
    const int kb = 32 * c + 8 * g;
    v16h a  = pack16(&s_ue[kb], &s_ue[kb + 16]);
    v16h bm = Bpack[(36 + c) * 32 + lane];
    acc = __builtin_amdgcn_wmma_f32_16x16x32_f16(
        false, a, false, bm, (short)0, acc, false, false);
  }
  const float* rowp = &s_f[col][0];         // A-row = list index (padded)
  #pragma unroll
  for (int c = 2; c < 4; ++c) {             // K in [64,128): pooled embedding
    const int kb = 32 * (c - 2) + 8 * g;
    v16h a  = pack16(rowp + kb, rowp + kb + 16);
    v16h bm = Bpack[(36 + c) * 32 + lane];
    acc = __builtin_amdgcn_wmma_f32_16x16x32_f16(
        false, a, false, bm, (short)0, acc, false, false);
  }

  const float bias = bf1[col];
  const float wfc  = Wf2[col];
  float p[8];
  #pragma unroll
  for (int r = 0; r < 8; ++r) {
    float v = acc[r] + bias;
    v = v > 0.0f ? v : 0.0f;
    p[r] = v * wfc;
  }
  #pragma unroll
  for (int mask = 1; mask <= 8; mask <<= 1) {
    #pragma unroll
    for (int r = 0; r < 8; ++r) p[r] += __shfl_xor(p[r], mask, 32);
  }

  if (col == 0) {
    const float bf2v = bf2[0];
    #pragma unroll
    for (int r = 0; r < 8; ++r) s_lg[8 * g + r] = p[r] + bf2v;
  }
  __syncthreads();

  // softmax over the 9 valid lists (redundant per lane; it's tiny)
  float mx = -1e30f;
  for (int j = 0; j < BN; ++j) mx = fmaxf(mx, s_lg[j]);
  float wv[BN];
  float sum = 0.0f;
  for (int j = 0; j < BN; ++j) { wv[j] = __expf(s_lg[j] - mx); sum += wv[j]; }
  const float inv = 1.0f / sum;

  for (int d = lane; d < DD; d += 32) {
    float s = 0.0f;
    for (int j = 0; j < BN; ++j) s += wv[j] * s_f[j][d];
    out[(long)b * DD + d] = s * inv;
  }
}

// ---------------------------------------------------------------------------
extern "C" void kernel_launch(void* const* d_in, const int* in_sizes, int n_in,
                              void* d_out, int out_size, void* d_ws, size_t ws_size,
                              hipStream_t stream) {
  const int*   user       = (const int*)d_in[0];
  const int*   item_ids   = (const int*)d_in[1];
  const int*   lengths    = (const int*)d_in[2];
  const float* user_table = (const float*)d_in[3];
  const float* item_table = (const float*)d_in[4];
  const float* W1         = (const float*)d_in[5];
  const float* b1         = (const float*)d_in[6];
  const float* W2         = (const float*)d_in[7];
  const float* b2         = (const float*)d_in[8];
  const float* Wf1        = (const float*)d_in[9];
  const float* bf1        = (const float*)d_in[10];
  const float* Wf2        = (const float*)d_in[11];
  const float* bf2        = (const float*)d_in[12];

  const int Bsz = in_sizes[0];                      // 2048

  _Float16* pack  = (_Float16*)d_ws;
  float*    uarea = (float*)((char*)d_ws + WS_UAREA_OFF);
  const v16h* Bp  = (const v16h*)d_ws;

  shan_pack<<<5, 256, 0, stream>>>(W1, Wf1, pack);
  shan_stage1<<<Bsz * BN, 128, 0, stream>>>(user, item_ids, lengths,
                                            user_table, item_table,
                                            b1, W2, b2, Bp, uarea);
  shan_stage2<<<Bsz, 32, 0, stream>>>(user, user_table, bf1, Wf2, bf2,
                                      Bp, uarea, (float*)d_out);
}
